// CDA_45389214384615
// MI455X (gfx1250) — compile-verified, matrix operand order checked by reading, not run on hardware
//
#include <hip/hip_runtime.h>
#include <math.h>

// ---------------------------------------------------------------------------
// Differential attention, MI455X (gfx1250), wave32, bf16 WMMA 16x16x32.
// Flash-style single pass over keys: no 2304x2304 score materialization.
// Round 2: async global->LDS double buffering in the attention kernel
// (guarded by __has_builtin), raw v_exp/v_rcp/v_rsq for the softmax math.
// ---------------------------------------------------------------------------

typedef __bf16 bf16;
typedef __attribute__((ext_vector_type(16))) __bf16 bf16x16;
typedef __attribute__((ext_vector_type(8)))  __bf16 bf16x8;
typedef __attribute__((ext_vector_type(8)))  float  f32x8;
typedef __attribute__((ext_vector_type(4)))  int    v4i;

struct bf16x8x2 { bf16x8 lo; bf16x8 hi; };

#define NH 8
#define HD 32
#define DD 64          // 2*hd
#define CC 256
#define NN 2304        // 48*48
#define KK2 512        // 2*C

static constexpr float LAMBDA_INIT = 0.3555090675909693f;     // 0.8 - 0.6*exp(-0.3)
static constexpr float OUT_SCALE   = 0.6444909324090307f;     // 1 - LAMBDA_INIT

#ifndef __has_builtin
#define __has_builtin(x) 0
#endif
#if __has_builtin(__builtin_amdgcn_global_load_async_to_lds_b128) && \
    __has_builtin(__builtin_amdgcn_s_wait_asynccnt)
#define USE_ASYNC_LDS 1
#else
#define USE_ASYNC_LDS 0
#endif

// ---------------------------------------------------------------------------
// Load a 16x32 operand (A layout, lane = M row; identical per-lane pattern
// serves B operands when the source is stored N-major, i.e. B^T row-major).
// Lane 0-15 : row = lane,    K = 0..7 (V0-3)  and 16..23 (V4-7)
// Lane 16-31: row = lane-16, K = 8..15        and 24..31
// ---------------------------------------------------------------------------
__device__ __forceinline__ bf16x16 load_tile16x32(const bf16* base, int rowStride) {
  const int lane = threadIdx.x & 31;
  const bf16* p = base + (size_t)(lane & 15) * rowStride + ((lane >> 4) << 3);
  bf16x8x2 pr;
  pr.lo = *(const bf16x8*)p;
  pr.hi = *(const bf16x8*)(p + 16);
  return __builtin_bit_cast(bf16x16, pr);
}

__device__ __forceinline__ f32x8 wmma_bf16(bf16x16 a, bf16x16 b, f32x8 c) {
  return __builtin_amdgcn_wmma_f32_16x16x32_bf16(false, a, false, b, (short)0, c,
                                                 false, false);
}

__device__ __forceinline__ bf16x8 cvt8(f32x8 v) {
  bf16x8 r;
#pragma unroll
  for (int i = 0; i < 8; ++i) r[i] = (bf16)v[i];
  return r;
}

__device__ __forceinline__ float grp16_max(float v) {
#pragma unroll
  for (int m = 8; m >= 1; m >>= 1) v = fmaxf(v, __shfl_xor(v, m, 32));
  return v;
}
__device__ __forceinline__ float grp16_sum(float v) {
#pragma unroll
  for (int m = 8; m >= 1; m >>= 1) v += __shfl_xor(v, m, 32);
  return v;
}
__device__ __forceinline__ float wave_sum(float v) {
#pragma unroll
  for (int m = 16; m >= 1; m >>= 1) v += __shfl_xor(v, m, 32);
  return v;
}

#if USE_ASYNC_LDS
// One 16-byte async DMA transfer, global -> LDS (tracked by ASYNCcnt).
// AS3 pointer = low 32 bits of the generic LDS address (aperture rule).
__device__ __forceinline__ void async_cp16(bf16* lds, const bf16* g) {
  __builtin_amdgcn_global_load_async_to_lds_b128(
      (v4i __attribute__((address_space(1)))*)(unsigned long long)g,
      (v4i __attribute__((address_space(3)))*)(unsigned int)(unsigned long long)lds,
      0, 0);
}
// Issue one K tile (32x64) + one V^T tile (64x32): 16 async b128 per lane.
__device__ __forceinline__ void stage_async(bf16* kl, bf16* vl,
                                            const bf16* gk, const bf16* gv,
                                            int lane) {
#pragma unroll
  for (int it = 0; it < 8; ++it) {
    int c = it * 32 + lane;            // 256 chunks of 16B each
    async_cp16(kl + c * 8, gk + c * 8);
    int d = c >> 2, part = c & 3;
    async_cp16(vl + c * 8, gv + (size_t)d * NN + part * 8);
  }
}
#endif

// ---------------------------------------------------------------------------
// Elementwise f32 -> bf16 (weights). Scale folds hd^-0.5 * log2(e) into q_w.
// ---------------------------------------------------------------------------
__global__ void k_cvt(const float* __restrict__ in, bf16* __restrict__ out,
                      int n, float s) {
  int i = blockIdx.x * blockDim.x + threadIdx.x;
  if (i < n) out[i] = (bf16)(in[i] * s);
}

// x (B,C,N) f32  ->  xt (B,N,C) bf16   (B^T layout for projection GEMM)
__global__ void k_xt(const float* __restrict__ x, bf16* __restrict__ xt) {
  int i = blockIdx.x * blockDim.x + threadIdx.x;   // B*N*C threads exactly
  int c = i % CC;
  int t = i / CC;
  int n = t % NN;
  int b = t / NN;
  xt[i] = (bf16)x[((size_t)b * CC + c) * NN + n];
}

// ---------------------------------------------------------------------------
// QKV projection: T = W(512x256) @ x(256x2304) per batch, one wave = 32x32 tile.
// mat 0/1 (Q/K): write transposed -> [b][h][n][d]; mat 2 (V): write [b][h][d][n].
// ---------------------------------------------------------------------------
__global__ void __launch_bounds__(32) k_proj(
    const bf16* __restrict__ wq, const bf16* __restrict__ wk,
    const bf16* __restrict__ wv, const bf16* __restrict__ xt,
    bf16* __restrict__ Q, bf16* __restrict__ K, bf16* __restrict__ Vt) {
  const int nt  = blockIdx.x;            // 0..71
  const int mt  = blockIdx.y;            // 0..15
  const int mat = blockIdx.z >> 1;       // 0..2
  const int b   = blockIdx.z & 1;
  const bf16* W = (mat == 0) ? wq : (mat == 1) ? wk : wv;
  const int m0 = mt * 32, n0 = nt * 32;
  const int lane = threadIdx.x & 31;

  f32x8 acc[2][2] = {};
  for (int k0 = 0; k0 < CC; k0 += 32) {
    bf16x16 a0 = load_tile16x32(W + (size_t)m0 * CC + k0, CC);
    bf16x16 a1 = load_tile16x32(W + (size_t)(m0 + 16) * CC + k0, CC);
    bf16x16 b0 = load_tile16x32(xt + ((size_t)b * NN + n0) * CC + k0, CC);
    bf16x16 b1 = load_tile16x32(xt + ((size_t)b * NN + n0 + 16) * CC + k0, CC);
    acc[0][0] = wmma_bf16(a0, b0, acc[0][0]);
    acc[0][1] = wmma_bf16(a0, b1, acc[0][1]);
    acc[1][0] = wmma_bf16(a1, b0, acc[1][0]);
    acc[1][1] = wmma_bf16(a1, b1, acc[1][1]);
  }

  const int h  = m0 >> 6;     // head (tile never crosses a 64-row head boundary)
  const int d0 = m0 & 63;
  if (mat < 2) {
    bf16* dst = (mat == 0) ? Q : K;
#pragma unroll
    for (int mi = 0; mi < 2; ++mi)
#pragma unroll
      for (int ni = 0; ni < 2; ++ni) {
        bf16x8 pk = cvt8(acc[mi][ni]);
        int n = n0 + ni * 16 + (lane & 15);
        int d = d0 + mi * 16 + ((lane >> 4) << 3);       // 8 consecutive d's
        *(bf16x8*)(dst + (((size_t)b * NH + h) * NN + n) * DD + d) = pk;
      }
  } else {
#pragma unroll
    for (int mi = 0; mi < 2; ++mi)
#pragma unroll
      for (int ni = 0; ni < 2; ++ni) {
        int n = n0 + ni * 16 + (lane & 15);
#pragma unroll
        for (int r = 0; r < 8; ++r) {
          int d = d0 + mi * 16 + ((lane >> 4) << 3) + r;
          Vt[(((size_t)b * NH + h) * DD + d) * NN + n] = (bf16)acc[mi][ni][r];
        }
      }
  }
}

// ---------------------------------------------------------------------------
// Flash differential attention. One wave per (b, h, 16 query rows).
// Streams over key tiles of 32 with online softmax for attn1 and attn2,
// then fused lambda-combine + RMS norm + (1-lambda) scaling.
// Q was pre-scaled by hd^-0.5 * log2(e), so v_exp_f32 == softmax exp.
// K/V tiles are double-buffered through LDS with async DMA when available.
// ---------------------------------------------------------------------------
__global__ void __launch_bounds__(32) k_attn(
    const bf16* __restrict__ Q, const bf16* __restrict__ K,
    const bf16* __restrict__ Vt,
    const float* __restrict__ lq1, const float* __restrict__ lk1,
    const float* __restrict__ lq2, const float* __restrict__ lk2,
    const float* __restrict__ rms_scale, bf16* __restrict__ O) {
#if USE_ASYNC_LDS
  __shared__ __align__(16) bf16 KL[2][32 * 64];   // key tile, [n][d]
  __shared__ __align__(16) bf16 VL[2][64 * 32];   // value tile transposed, [d][n]
#else
  __shared__ __align__(16) bf16 KL[1][32 * 64];
  __shared__ __align__(16) bf16 VL[1][64 * 32];
#endif
  __shared__ __align__(16) bf16 PL1[16 * 32];     // probs stream 1, [m][j]
  __shared__ __align__(16) bf16 PL2[16 * 32];

  const int mt = blockIdx.x;   // 0..143
  const int h  = blockIdx.y;   // 0..7
  const int b  = blockIdx.z;   // 0..1
  const int lane = threadIdx.x & 31;
  const int m0 = mt * 16;

  // lambda for this head
  float pr1 = lq1[h * HD + lane] * lk1[h * HD + lane];
  float pr2 = lq2[h * HD + lane] * lk2[h * HD + lane];
  const float lam = __expf(wave_sum(pr1)) - __expf(wave_sum(pr2)) + LAMBDA_INIT;

  const size_t bh = (size_t)b * NH + h;
  const bf16* Qb = Q + (bh * NN + m0) * DD;
  const bf16x16 q1 = load_tile16x32(Qb, DD);        // d = 0..31
  const bf16x16 q2 = load_tile16x32(Qb + 32, DD);   // d = 32..63

  const bf16* Kbase = K + bh * NN * DD;             // key row n -> stride DD
  const bf16* Vbase = Vt + bh * (size_t)DD * NN;    // V^T row d -> stride NN

  float rs[4];
#pragma unroll
  for (int t = 0; t < 4; ++t) rs[t] = rms_scale[t * 16 + (lane & 15)];

  f32x8 o1[4] = {}, o2[4] = {};
  float m1[8], l1[8], m2[8], l2[8];
#pragma unroll
  for (int r = 0; r < 8; ++r) {
    m1[r] = -__builtin_inff(); l1[r] = 0.f;
    m2[r] = -__builtin_inff(); l2[r] = 0.f;
  }

#if USE_ASYNC_LDS
  stage_async(KL[0], VL[0], Kbase, Vbase, lane);    // prime tile 0
#endif

  for (int n0 = 0, buf = 0; n0 < NN; n0 += 32, buf ^= 1) {
#if USE_ASYNC_LDS
    // Prefetch next tile into the other buffer, then wait for the current one.
    // Async loads complete in order, so ASYNCcnt <= 16 => current tile landed.
    if (n0 + 32 < NN) {
      stage_async(KL[buf ^ 1], VL[buf ^ 1],
                  Kbase + (size_t)(n0 + 32) * DD, Vbase + (n0 + 32), lane);
      __builtin_amdgcn_s_wait_asynccnt(16);
    } else {
      __builtin_amdgcn_s_wait_asynccnt(0);
    }
    asm volatile("" ::: "memory");
    const bf16* KLc = KL[buf];
    const bf16* VLc = VL[buf];
#else
    if (n0 + 32 < NN) {   // speculative prefetch of the next tile
      __builtin_prefetch(Kbase + (size_t)(n0 + 32) * DD + lane * 8, 0, 1);
      __builtin_prefetch(Vbase + (n0 + 32) + (size_t)lane * NN, 0, 1);
      __builtin_prefetch(Vbase + (n0 + 32) + (size_t)(lane + 32) * NN, 0, 1);
    }
    {  // synchronous staging fallback: coalesced b128 copies through VGPRs
      const uint4* gk = (const uint4*)(Kbase + (size_t)n0 * DD);
      uint4* sk = (uint4*)KL[0];
#pragma unroll
      for (int it = 0; it < 8; ++it) sk[it * 32 + lane] = gk[it * 32 + lane];
      const bf16* gv = Vbase + n0;
      uint4* sv = (uint4*)VL[0];
#pragma unroll
      for (int it = 0; it < 8; ++it) {
        int c = it * 32 + lane;
        int d = c >> 2, part = c & 3;
        sv[c] = *(const uint4*)(gv + (size_t)d * NN + part * 8);
      }
    }
    const bf16* KLc = KL[0];
    const bf16* VLc = VL[0];
#endif

    // Scores: one WMMA per 16x16 tile (K = hd = 32).
    const f32x8 zz = {0.f, 0.f, 0.f, 0.f, 0.f, 0.f, 0.f, 0.f};
    bf16x16 kb0 = load_tile16x32(KLc + 0 * 64, 64);        // keys 0-15,  d 0-31
    bf16x16 kb1 = load_tile16x32(KLc + 16 * 64, 64);       // keys 16-31, d 0-31
    bf16x16 kb2 = load_tile16x32(KLc + 0 * 64 + 32, 64);   // keys 0-15,  d 32-63
    bf16x16 kb3 = load_tile16x32(KLc + 16 * 64 + 32, 64);  // keys 16-31, d 32-63
    f32x8 s10 = wmma_bf16(q1, kb0, zz);
    f32x8 s11 = wmma_bf16(q1, kb1, zz);
    f32x8 s20 = wmma_bf16(q2, kb2, zz);
    f32x8 s21 = wmma_bf16(q2, kb3, zz);

    // Online softmax per row (row r lives in VGPR r of each half-wave).
    float sf1[8], sf2[8];
#pragma unroll
    for (int r = 0; r < 8; ++r) {
      const int row = r + ((lane >> 4) << 3);
      const int col = lane & 15;
      {
        float rm = grp16_max(fmaxf(s10[r], s11[r]));
        float mn = fmaxf(m1[r], rm);
        float p0 = __builtin_amdgcn_exp2f(s10[r] - mn);
        float p1 = __builtin_amdgcn_exp2f(s11[r] - mn);
        float sf = __builtin_amdgcn_exp2f(m1[r] - mn);
        l1[r] = l1[r] * sf + grp16_sum(p0 + p1);
        m1[r] = mn; sf1[r] = sf;
        PL1[row * 32 + col]      = (bf16)p0;
        PL1[row * 32 + 16 + col] = (bf16)p1;
      }
      {
        float rm = grp16_max(fmaxf(s20[r], s21[r]));
        float mn = fmaxf(m2[r], rm);
        float p0 = __builtin_amdgcn_exp2f(s20[r] - mn);
        float p1 = __builtin_amdgcn_exp2f(s21[r] - mn);
        float sf = __builtin_amdgcn_exp2f(m2[r] - mn);
        l2[r] = l2[r] * sf + grp16_sum(p0 + p1);
        m2[r] = mn; sf2[r] = sf;
        PL2[row * 32 + col]      = (bf16)p0;
        PL2[row * 32 + 16 + col] = (bf16)p1;
      }
    }

    // Rescale accumulators, then O += P @ V (K = 32 keys per step).
#pragma unroll
    for (int t = 0; t < 4; ++t)
#pragma unroll
      for (int r = 0; r < 8; ++r) { o1[t][r] *= sf1[r]; o2[t][r] *= sf2[r]; }

    bf16x16 pa1 = load_tile16x32(PL1, 32);   // LDS round-trip = transpose to A layout
    bf16x16 pa2 = load_tile16x32(PL2, 32);
#pragma unroll
    for (int t = 0; t < 4; ++t) {
      bf16x16 vb = load_tile16x32(VLc + t * 16 * 32, 32);  // out cols t*16..+15
      o1[t] = wmma_bf16(pa1, vb, o1[t]);
      o2[t] = wmma_bf16(pa2, vb, o2[t]);
    }
  }

  // Finalize: combine streams, RMS-norm over d=64, write O^T[b][n][h*64+d].
#pragma unroll
  for (int r = 0; r < 8; ++r) {
    float inv1 = __builtin_amdgcn_rcpf(l1[r]);
    float inv2 = lam * __builtin_amdgcn_rcpf(l2[r]);
    float c[4];
    float ss = 0.f;
#pragma unroll
    for (int t = 0; t < 4; ++t) {
      c[t] = o1[t][r] * inv1 - inv2 * o2[t][r];
      ss += c[t] * c[t];
    }
    ss = grp16_sum(ss);
    float g = __builtin_amdgcn_rsqf(ss * (1.0f / 64.0f) + 1e-5f) * OUT_SCALE;
    int n = m0 + r + ((lane >> 4) << 3);
    bf16* dst = O + ((size_t)b * NN + n) * KK2 + h * 64;
#pragma unroll
    for (int t = 0; t < 4; ++t)
      dst[t * 16 + (lane & 15)] = (bf16)(c[t] * g * rs[t]);
  }
}

// ---------------------------------------------------------------------------
// Output projection: y = out_w(256x512) @ O(512x2304), fp32 result to d_out.
// ---------------------------------------------------------------------------
__global__ void __launch_bounds__(32) k_out(
    const bf16* __restrict__ wo, const bf16* __restrict__ Ot,
    float* __restrict__ y) {
  const int n0 = blockIdx.x * 32;
  const int m0 = blockIdx.y * 32;
  const int b  = blockIdx.z;
  const int lane = threadIdx.x & 31;

  f32x8 acc[2][2] = {};
  for (int k0 = 0; k0 < KK2; k0 += 32) {
    bf16x16 a0 = load_tile16x32(wo + (size_t)m0 * KK2 + k0, KK2);
    bf16x16 a1 = load_tile16x32(wo + (size_t)(m0 + 16) * KK2 + k0, KK2);
    bf16x16 b0 = load_tile16x32(Ot + ((size_t)b * NN + n0) * KK2 + k0, KK2);
    bf16x16 b1 = load_tile16x32(Ot + ((size_t)b * NN + n0 + 16) * KK2 + k0, KK2);
    acc[0][0] = wmma_bf16(a0, b0, acc[0][0]);
    acc[0][1] = wmma_bf16(a0, b1, acc[0][1]);
    acc[1][0] = wmma_bf16(a1, b0, acc[1][0]);
    acc[1][1] = wmma_bf16(a1, b1, acc[1][1]);
  }
#pragma unroll
  for (int mi = 0; mi < 2; ++mi)
#pragma unroll
    for (int ni = 0; ni < 2; ++ni) {
      int n = n0 + ni * 16 + (lane & 15);
#pragma unroll
      for (int r = 0; r < 8; ++r) {
        int m = m0 + mi * 16 + ((lane >> 4) << 3) + r;
        y[((size_t)b * CC + m) * NN + n] = acc[mi][ni][r];
      }
    }
}

// ---------------------------------------------------------------------------
extern "C" void kernel_launch(void* const* d_in, const int* in_sizes, int n_in,
                              void* d_out, int out_size, void* d_ws, size_t ws_size,
                              hipStream_t stream) {
  (void)in_sizes; (void)n_in; (void)out_size; (void)ws_size;
  const float* x   = (const float*)d_in[0];
  const float* qw  = (const float*)d_in[1];
  const float* kw  = (const float*)d_in[2];
  const float* vw  = (const float*)d_in[3];
  const float* ow  = (const float*)d_in[4];
  const float* lq1 = (const float*)d_in[5];
  const float* lk1 = (const float*)d_in[6];
  const float* lq2 = (const float*)d_in[7];
  const float* lk2 = (const float*)d_in[8];
  const float* rss = (const float*)d_in[9];
  float* y = (float*)d_out;

  // Workspace layout (bf16), total 22,282,240 bytes.
  bf16* xt  = (bf16*)d_ws;                 // 2*2304*256
  bf16* wqb = xt + 2 * NN * CC;            // 512*256 each
  bf16* wkb = wqb + KK2 * CC;
  bf16* wvb = wkb + KK2 * CC;
  bf16* wob = wvb + KK2 * CC;              // 256*512
  bf16* Qb  = wob + CC * KK2;              // 2*8*2304*64 each
  bf16* Kb  = Qb + 2 * NH * NN * DD;
  bf16* Vtb = Kb + 2 * NH * NN * DD;
  bf16* Ob  = Vtb + 2 * NH * NN * DD;      // 2*2304*512

  const float qscale = 0.17677669529663687f * 1.4426950408889634f; // hd^-.5 * log2e
  k_cvt<<<512, 256, 0, stream>>>(qw, wqb, KK2 * CC, qscale);
  k_cvt<<<512, 256, 0, stream>>>(kw, wkb, KK2 * CC, 1.0f);
  k_cvt<<<512, 256, 0, stream>>>(vw, wvb, KK2 * CC, 1.0f);
  k_cvt<<<512, 256, 0, stream>>>(ow, wob, CC * KK2, 1.0f);
  k_xt<<<4608, 256, 0, stream>>>(x, xt);

  dim3 gp(NN / 32, KK2 / 32, 6);           // 72 x 16 x (3 mats * 2 batches)
  k_proj<<<gp, 32, 0, stream>>>(wqb, wkb, wvb, xt, Qb, Kb, Vtb);

  dim3 ga(NN / 16, NH, 2);                 // 144 x 8 x 2
  k_attn<<<ga, 32, 0, stream>>>(Qb, Kb, Vtb, lq1, lk1, lq2, lk2, rss, Ob);

  dim3 go(NN / 32, CC / 32, 2);            // 72 x 8 x 2
  k_out<<<go, 32, 0, stream>>>(wob, Ob, y);
}
